// DecoderWaveFunction_29618094473916
// MI455X (gfx1250) — matrix operand order repaired and checked by reading, not compiled, not run on hardware
//
#include <hip/hip_runtime.h>

typedef __attribute__((ext_vector_type(16))) _Float16 v16h;
typedef __attribute__((ext_vector_type(8)))  _Float16 v8h;
typedef __attribute__((ext_vector_type(8)))  float    v8f;

#define NEGBIG (-1.0e30f)

// ---------------- LDS layout (dynamic) ----------------
// Hs : float    [256][64]   residual stream            65536 B
// As : _Float16 [256][64]   ln-out / attention-out     32768 B
// Ss : _Float16 [256][192]  qkv scratch / ff chunk     98304 B
// Ws : _Float16 weights (transposed, N x K)            32768 B
// rowlp float[256], amps float[8], phb float[8][64], phb2 float[8][64]
#define SM_H   0
#define SM_A   65536
#define SM_S   98304
#define SM_W   196608
#define SM_LP  229376
#define SM_AMP 230400
#define SM_PH1 230432
#define SM_PH2 232480
#define SMEM_BYTES 234528

union U16frag { v16h v; v8h h[2]; };

// A fragment, 16-bit A 16x32 layout (contiguous, matches B):
// lane l: row = m0 + (l&15); halves 0-15 -> K = kb..kb+15
// with kb = k0 + 16*(l>>4)
__device__ __forceinline__ v16h load_fragA(const _Float16* A, int lda, int m0, int k0, int lane) {
  int r  = m0 + (lane & 15);
  int kb = k0 + ((lane >> 4) << 4);
  return *(const v16h*)(A + r * lda + kb);
}

// B fragment from transposed weights (Bt is N x K row-major):
// lane l: col = n0 + (l&15); halves 0-15 -> K = k0 + 16*(l>>4) .. +15  (contiguous)
__device__ __forceinline__ v16h load_fragB(const _Float16* Bt, int ldb, int n0, int k0, int lane) {
  int c  = n0 + (lane & 15);
  int kb = k0 + ((lane >> 4) << 4);
  return *(const v16h*)(Bt + c * ldb + kb);
}

// C/D tile epilogues (16x16 f32 accumulator layout: lane = col | (rowhalf<<4))
__device__ __forceinline__ void store_tile_f16(v8f acc, int m0, int n0,
                                               const float* bias, _Float16* C,
                                               int ldc, bool relu, int lane) {
  int col = n0 + (lane & 15);
  int rb  = m0 + ((lane >> 4) << 3);
  float bb = bias ? bias[col] : 0.0f;
#pragma unroll
  for (int r = 0; r < 8; ++r) {
    float v = acc[r] + bb;
    if (relu) v = fmaxf(v, 0.0f);
    C[(rb + r) * ldc + col] = (_Float16)v;
  }
}

__device__ __forceinline__ void add_tile_f32(v8f acc, int m0, int n0,
                                             const float* bias, float* H,
                                             int ldh, int lane) {
  int col = n0 + (lane & 15);
  int rb  = m0 + ((lane >> 4) << 3);
  float bb = bias ? bias[col] : 0.0f;
#pragma unroll
  for (int r = 0; r < 8; ++r) {
    H[(rb + r) * ldh + col] += acc[r] + bb;
  }
}

#define WMMA_F16(A_, B_, C_) \
  __builtin_amdgcn_wmma_f32_16x16x32_f16(false, (A_), false, (B_), (short)0, (C_), false, false)

// C = A(MxK,f16 LDS) * B(KxN via Bt NxK f16 LDS) + bias, optional relu, store f16
// 2x2 register blocking: each wave owns a 32x32 output block per iteration.
__device__ __forceinline__ void gemm_f16out(const _Float16* A, int lda,
                                            const _Float16* Bt, int ldb,
                                            int M, int N, int K,
                                            const float* bias,
                                            _Float16* C, int ldc,
                                            bool relu, int tid) {
  int wave = tid >> 5, lane = tid & 31;
  int ntn2 = N >> 5;
  int T = (M >> 5) * ntn2;
  for (int t = wave; t < T; t += 8) {
    int m0 = (t / ntn2) << 5, n0 = (t % ntn2) << 5;
    v8f a00 = {0.f,0.f,0.f,0.f,0.f,0.f,0.f,0.f};
    v8f a01 = a00, a10 = a00, a11 = a00;
    for (int k0 = 0; k0 < K; k0 += 32) {
      v16h fa0 = load_fragA(A, lda, m0,      k0, lane);
      v16h fa1 = load_fragA(A, lda, m0 + 16, k0, lane);
      v16h fb0 = load_fragB(Bt, ldb, n0,      k0, lane);
      v16h fb1 = load_fragB(Bt, ldb, n0 + 16, k0, lane);
      a00 = WMMA_F16(fa0, fb0, a00);
      a01 = WMMA_F16(fa0, fb1, a01);
      a10 = WMMA_F16(fa1, fb0, a10);
      a11 = WMMA_F16(fa1, fb1, a11);
    }
    store_tile_f16(a00, m0,      n0,      bias, C, ldc, relu, lane);
    store_tile_f16(a01, m0,      n0 + 16, bias, C, ldc, relu, lane);
    store_tile_f16(a10, m0 + 16, n0,      bias, C, ldc, relu, lane);
    store_tile_f16(a11, m0 + 16, n0 + 16, bias, C, ldc, relu, lane);
  }
}

// H(f32 LDS) += A*B + bias   (residual accumulate), same 2x2 blocking
__device__ __forceinline__ void gemm_addf32(const _Float16* A, int lda,
                                            const _Float16* Bt, int ldb,
                                            int M, int N, int K,
                                            const float* bias,
                                            float* H, int ldh, int tid) {
  int wave = tid >> 5, lane = tid & 31;
  int ntn2 = N >> 5;
  int T = (M >> 5) * ntn2;
  for (int t = wave; t < T; t += 8) {
    int m0 = (t / ntn2) << 5, n0 = (t % ntn2) << 5;
    v8f a00 = {0.f,0.f,0.f,0.f,0.f,0.f,0.f,0.f};
    v8f a01 = a00, a10 = a00, a11 = a00;
    for (int k0 = 0; k0 < K; k0 += 32) {
      v16h fa0 = load_fragA(A, lda, m0,      k0, lane);
      v16h fa1 = load_fragA(A, lda, m0 + 16, k0, lane);
      v16h fb0 = load_fragB(Bt, ldb, n0,      k0, lane);
      v16h fb1 = load_fragB(Bt, ldb, n0 + 16, k0, lane);
      a00 = WMMA_F16(fa0, fb0, a00);
      a01 = WMMA_F16(fa0, fb1, a01);
      a10 = WMMA_F16(fa1, fb0, a10);
      a11 = WMMA_F16(fa1, fb1, a11);
    }
    add_tile_f32(a00, m0,      n0,      bias, H, ldh, lane);
    add_tile_f32(a01, m0,      n0 + 16, bias, H, ldh, lane);
    add_tile_f32(a10, m0 + 16, n0,      bias, H, ldh, lane);
    add_tile_f32(a11, m0 + 16, n0 + 16, bias, H, ldh, lane);
  }
}

// layernorm one row per thread (256 rows), out f16
__device__ __forceinline__ void layernorm256(const float* Hs, const float* g,
                                             const float* b, _Float16* out, int tid) {
  const float* hr = Hs + tid * 64;
  float m = 0.f;
#pragma unroll
  for (int d = 0; d < 64; ++d) m += hr[d];
  m *= (1.f / 64.f);
  float var = 0.f;
#pragma unroll
  for (int d = 0; d < 64; ++d) { float t = hr[d] - m; var += t * t; }
  var *= (1.f / 64.f);
  float rs = rsqrtf(var + 1e-5f);
  _Float16* orow = out + tid * 64;
#pragma unroll
  for (int d = 0; d < 64; ++d)
    orow[d] = (_Float16)((hr[d] - m) * rs * g[d] + b[d]);
}

// stage W (K x N row-major f32 global) -> out (N x K row-major f16 LDS)
__device__ __forceinline__ void stageT(const float* W, int K, int N,
                                       _Float16* out, int tid) {
  int total = K * N;
  for (int i = tid; i < total; i += 256) {
    int n = i / K, k = i - n * K;
    out[i] = (_Float16)W[k * N + n];
  }
}

__global__ __launch_bounds__(256) void dwf_kernel(
    const float* __restrict__ x,      const float* __restrict__ tok_emb,
    const float* __restrict__ pos_emb,const float* __restrict__ ln1_g,
    const float* __restrict__ ln1_b,  const float* __restrict__ wqkv,
    const float* __restrict__ bqkv,   const float* __restrict__ wo,
    const float* __restrict__ bo,     const float* __restrict__ ln2_g,
    const float* __restrict__ ln2_b,  const float* __restrict__ w1,
    const float* __restrict__ b1,     const float* __restrict__ w2,
    const float* __restrict__ b2,     const float* __restrict__ lnf_g,
    const float* __restrict__ lnf_b,  const float* __restrict__ w_out,
    const float* __restrict__ b_out,  const float* __restrict__ wp1,
    const float* __restrict__ bp1,    const float* __restrict__ wp2,
    const float* __restrict__ bp2,    const float* __restrict__ wp3,
    const float* __restrict__ bp3,    float* __restrict__ out) {
  extern __shared__ __align__(128) char smem[];
  float*    Hs    = (float*)(smem + SM_H);
  _Float16* As    = (_Float16*)(smem + SM_A);
  _Float16* Ss    = (_Float16*)(smem + SM_S);
  _Float16* Ws    = (_Float16*)(smem + SM_W);
  float*    rowlp = (float*)(smem + SM_LP);
  float*    amps  = (float*)(smem + SM_AMP);
  float*    phb   = (float*)(smem + SM_PH1);
  float*    phb2  = (float*)(smem + SM_PH2);

  const int tid = threadIdx.x;

  // ---- token + positional embedding: one row per thread ----
  {
    int s = tid >> 5, t = tid & 31;
    long gs = (long)blockIdx.x * 8 + s;
    int tok = 4;
    if (t != 0) {
      const float* xs = x + gs * 64;
      int ub = xs[2 * (t - 1)]     > 0.f;
      int db = xs[2 * (t - 1) + 1] > 0.f;
      tok = ub + 2 * db;
    }
    float* hrow = Hs + tid * 64;
#pragma unroll
    for (int d = 0; d < 64; ++d)
      hrow[d] = tok_emb[tok * 64 + d] + pos_emb[t * 64 + d];
  }
  __syncthreads();

  // ---- transformer layers ----
  for (int l = 0; l < 6; ++l) {
    if (l + 1 < 6) __builtin_prefetch(wqkv + (long)(l + 1) * 64 * 192, 0, 1);

    // LN1 -> As ; stage wqkv^T (192x64)
    layernorm256(Hs, ln1_g + l * 64, ln1_b + l * 64, As, tid);
    stageT(wqkv + (long)l * 64 * 192, 64, 192, Ws, tid);
    __syncthreads();

    // qkv = As @ wqkv + bqkv  -> Ss (256 x 192)
    gemm_f16out(As, 64, Ws, 64, 256, 192, 64, bqkv + l * 192, Ss, 192, false, tid);
    __syncthreads();

    // ---- attention (VALU): task = (sample, head, query) ----
    for (int task = tid; task < 2048; task += 256) {
      int s  = task >> 8;
      int hh = (task >> 5) & 7;
      int qi = task & 31;
      int base = s << 5;
      const _Float16* qp = Ss + (base + qi) * 192 + hh * 8;
      float q[8];
#pragma unroll
      for (int d = 0; d < 8; ++d) q[d] = (float)qp[d] * 0.35355339059327373f;
      float sc[32];
      float mx = NEGBIG;
#pragma unroll
      for (int tt = 0; tt < 32; ++tt) {
        float dot = NEGBIG;
        if (tt <= qi) {
          const _Float16* kp = Ss + (base + tt) * 192 + 64 + hh * 8;
          dot = 0.f;
#pragma unroll
          for (int d = 0; d < 8; ++d) dot += q[d] * (float)kp[d];
        }
        sc[tt] = dot;
        mx = fmaxf(mx, dot);
      }
      float sum = 0.f;
#pragma unroll
      for (int tt = 0; tt < 32; ++tt) {
        float e = (tt <= qi) ? expf(sc[tt] - mx) : 0.f;
        sc[tt] = e;
        sum += e;
      }
      float inv = 1.0f / sum;
      float o[8] = {0.f,0.f,0.f,0.f,0.f,0.f,0.f,0.f};
#pragma unroll
      for (int tt = 0; tt < 32; ++tt) {
        if (tt <= qi) {
          float p = sc[tt] * inv;
          const _Float16* vp = Ss + (base + tt) * 192 + 128 + hh * 8;
#pragma unroll
          for (int d = 0; d < 8; ++d) o[d] += p * (float)vp[d];
        }
      }
      _Float16* op = As + (base + qi) * 64 + hh * 8;
#pragma unroll
      for (int d = 0; d < 8; ++d) op[d] = (_Float16)o[d];
    }
    // stage wo^T (64x64)
    stageT(wo + (long)l * 64 * 64, 64, 64, Ws, tid);
    __syncthreads();

    // h += attn_out @ wo + bo
    gemm_addf32(As, 64, Ws, 64, 256, 64, 64, bo + l * 64, Hs, 64, tid);
    __syncthreads();

    // LN2 -> As
    layernorm256(Hs, ln2_g + l * 64, ln2_b + l * 64, As, tid);
    __syncthreads();

    // MLP, K-split into two 128-wide hidden chunks
    const float* W1l = w1 + (long)l * 64 * 256;
    const float* W2l = w2 + (long)l * 256 * 64;
    for (int c = 0; c < 2; ++c) {
      // w1^T chunk: Ws[0..8191]  = w1[k][128c+n]   (n<128, k<64)
      for (int i = tid; i < 128 * 64; i += 256) {
        int n = i >> 6, k = i & 63;
        Ws[i] = (_Float16)W1l[k * 256 + (c << 7) + n];
      }
      // w2^T chunk: Ws[8192..]   = w2[128c+k][n]   (n<64, k<128)
      for (int i = tid; i < 64 * 128; i += 256) {
        int n = i >> 7, k = i & 127;
        Ws[8192 + i] = (_Float16)W2l[((c << 7) + k) * 64 + n];
      }
      __syncthreads();
      // ff = relu(As @ w1_c + b1_c)  -> Ss (256 x 128)
      gemm_f16out(As, 64, Ws, 64, 256, 128, 64, b1 + l * 256 + (c << 7), Ss, 128, true, tid);
      __syncthreads();
      // h += ff @ w2_c (+ b2 once)
      gemm_addf32(Ss, 128, Ws + 8192, 128, 256, 64, 128,
                  (c == 0) ? (b2 + l * 64) : nullptr, Hs, 64, tid);
      __syncthreads();
    }
  }

  // ---- final LN + logits + mask + log_softmax (one row per thread) ----
  {
    const float* hr = Hs + tid * 64;
    float m = 0.f;
#pragma unroll
    for (int d = 0; d < 64; ++d) m += hr[d];
    m *= (1.f / 64.f);
    float var = 0.f;
#pragma unroll
    for (int d = 0; d < 64; ++d) { float t2 = hr[d] - m; var += t2 * t2; }
    var *= (1.f / 64.f);
    float rs = rsqrtf(var + 1e-5f);
    float lg[4] = { b_out[0], b_out[1], b_out[2], b_out[3] };
#pragma unroll
    for (int d = 0; d < 64; ++d) {
      float ad = (hr[d] - m) * rs * lnf_g[d] + lnf_b[d];
#pragma unroll
      for (int c = 0; c < 4; ++c) lg[c] += ad * w_out[d * 4 + c];
    }
    int s = tid >> 5, t = tid & 31;
    long gs = (long)blockIdx.x * 8 + s;
    const float* xs = x + gs * 64;
    int nup = 0, ndn = 0;
    for (int j = 0; j < t; ++j) {
      nup += (xs[2 * j]     > 0.f);
      ndn += (xs[2 * j + 1] > 0.f);
    }
    int ub = xs[2 * t] > 0.f, db = xs[2 * t + 1] > 0.f;
    int pair = ub + 2 * db;
    int rem = 31 - t;
    const int du[4]  = {0, 1, 0, 1};
    const int dd2[4] = {0, 0, 1, 1};
    float msk[4];
    float mx = NEGBIG;
#pragma unroll
    for (int c = 0; c < 4; ++c) {
      int nu = nup + du[c], nd = ndn + dd2[c];
      bool ok = (nu <= 16) && (nd <= 16) && ((16 - nu) <= rem) && ((16 - nd) <= rem);
      msk[c] = ok ? lg[c] : NEGBIG;
      mx = fmaxf(mx, msk[c]);
    }
    float sum = 0.f;
#pragma unroll
    for (int c = 0; c < 4; ++c) sum += expf(msk[c] - mx);
    rowlp[tid] = msk[pair] - mx - logf(sum);
  }
  __syncthreads();
  if (tid < 8) {
    float a2 = 0.f;
    for (int t = 0; t < 32; ++t) a2 += rowlp[tid * 32 + t];
    amps[tid] = a2;
  }
  __syncthreads();

  // ---- phase head: wave w handles sample w ----
  {
    int wave = tid >> 5, lane = tid & 31;
    long gs = (long)blockIdx.x * 8 + wave;
    const float* xs = x + gs * 64;
#pragma unroll
    for (int rep = 0; rep < 2; ++rep) {
      int j = lane + rep * 32;
      float acc = bp1[j];
      for (int i = 0; i < 64; ++i) acc += xs[i] * wp1[i * 64 + j];
      phb[wave * 64 + j] = fmaxf(acc, 0.f);
    }
    __syncthreads();
#pragma unroll
    for (int rep = 0; rep < 2; ++rep) {
      int j = lane + rep * 32;
      float acc = bp2[j];
      for (int i = 0; i < 64; ++i) acc += phb[wave * 64 + i] * wp2[i * 64 + j];
      phb2[wave * 64 + j] = fmaxf(acc, 0.f);
    }
    __syncthreads();
    if (lane == 0) {
      float ph = bp3[0];
      for (int i = 0; i < 64; ++i) ph += phb2[wave * 64 + i] * wp3[i];
      float amp = expf(0.5f * amps[wave]);
      out[2 * gs]     = cosf(ph) * amp;
      out[2 * gs + 1] = sinf(ph) * amp;
    }
  }
}

extern "C" void kernel_launch(void* const* d_in, const int* in_sizes, int n_in,
                              void* d_out, int out_size, void* d_ws, size_t ws_size,
                              hipStream_t stream) {
  (void)n_in; (void)d_ws; (void)ws_size; (void)out_size;
  const float* x       = (const float*)d_in[0];
  const float* tok_emb = (const float*)d_in[1];
  const float* pos_emb = (const float*)d_in[2];
  const float* ln1_g   = (const float*)d_in[3];
  const float* ln1_b   = (const float*)d_in[4];
  const float* wqkv    = (const float*)d_in[5];
  const float* bqkv    = (const float*)d_in[6];
  const float* wo      = (const float*)d_in[7];
  const float* bo      = (const float*)d_in[8];
  const float* ln2_g   = (const float*)d_in[9];
  const float* ln2_b   = (const float*)d_in[10];
  const float* w1      = (const float*)d_in[11];
  const float* b1      = (const float*)d_in[12];
  const float* w2      = (const float*)d_in[13];
  const float* b2      = (const float*)d_in[14];
  const float* lnf_g   = (const float*)d_in[15];
  const float* lnf_b   = (const float*)d_in[16];
  const float* w_out   = (const float*)d_in[17];
  const float* b_out   = (const float*)d_in[18];
  const float* wp1     = (const float*)d_in[19];
  const float* bp1     = (const float*)d_in[20];
  const float* wp2     = (const float*)d_in[21];
  const float* bp2     = (const float*)d_in[22];
  const float* wp3     = (const float*)d_in[23];
  const float* bp3     = (const float*)d_in[24];

  int B = in_sizes[0] / 64;
  int grid = B / 8;
  dwf_kernel<<<grid, 256, SMEM_BYTES, stream>>>(
      x, tok_emb, pos_emb, ln1_g, ln1_b, wqkv, bqkv, wo, bo, ln2_g, ln2_b,
      w1, b1, w2, b2, lnf_g, lnf_b, w_out, b_out, wp1, bp1, wp2, bp2, wp3, bp3,
      (float*)d_out);
}
